// ExpertChoiceMoE_20796231647728
// MI455X (gfx1250) — matrix operand order repaired and checked by reading, not compiled, not run on hardware
//
#include <hip/hip_runtime.h>

// Problem constants (from reference): B=4, T=2048, C=1024, H=2048, E=8
#define NTOK 8192   // B*T
#define CDIM 1024
#define HDIM 2048
#define NEXP 8
#define CAP  1024   // capacity = NTOK/NEXP

// GEMM tiling
#define BM 128
#define BN 128
#define BK 32

typedef __attribute__((ext_vector_type(16))) __bf16    v16bf;
typedef __attribute__((ext_vector_type(8)))  float     v8f;
typedef __attribute__((ext_vector_type(4)))  unsigned  u32x4;
typedef __attribute__((ext_vector_type(2)))  unsigned  u32x2;
typedef __attribute__((ext_vector_type(4)))  float     f32x4;
typedef __attribute__((ext_vector_type(4)))  int       i32x4;

union FragB { u32x4 q[2]; v16bf v; };

__device__ __forceinline__ unsigned short f2bf(float f) {
  unsigned u = __builtin_bit_cast(unsigned, f);
  unsigned r = 0x7FFFu + ((u >> 16) & 1u);   // round-to-nearest-even
  return (unsigned short)((u + r) >> 16);
}
__device__ __forceinline__ unsigned pk2(float a, float b) {
  return (unsigned)f2bf(a) | ((unsigned)f2bf(b) << 16);
}

// Async global->LDS copy (CDNA5 ASYNCcnt path); compile-time guarded so the
// build cannot regress if this toolchain lacks the builtin.
// Probe-learned signature: (int4 addrspace(1)*, int4 addrspace(3)*, imm, imm)
#if __has_builtin(__builtin_amdgcn_global_load_async_to_lds_b128)
#define HAVE_ASYNC_LDS 1
__device__ __forceinline__ void async_copy16(const void* g, void* l) {
  __builtin_amdgcn_global_load_async_to_lds_b128(
      (__attribute__((address_space(1))) i32x4*)(g),
      (__attribute__((address_space(3))) i32x4*)(l), 0, 0);
}
__device__ __forceinline__ void wait_async0() {
  asm volatile("s_wait_asynccnt 0" ::: "memory");
}
#else
#define HAVE_ASYNC_LDS 0
#endif

// ---------------------------------------------------------------------------
// 1) Gate: scores[e][n] = dot(x[n], Wg[e])
// ---------------------------------------------------------------------------
__global__ __launch_bounds__(256) void gate_kernel(const float* __restrict__ x,
                                                   const float* __restrict__ Wg,
                                                   float* __restrict__ scores) {
  __shared__ float wg_s[NEXP * CDIM];   // 32 KB
  const int tid = threadIdx.x;
  for (int i = tid * 4; i < NEXP * CDIM; i += 1024)
    *(f32x4*)&wg_s[i] = *(const f32x4*)&Wg[i];
  __syncthreads();

  const int wid = tid >> 5, lane = tid & 31;
  const int n = blockIdx.x * 8 + wid;
  const float* xr = x + (size_t)n * CDIM;

  float acc[NEXP];
#pragma unroll
  for (int e = 0; e < NEXP; ++e) acc[e] = 0.f;
  for (int c = lane; c < CDIM; c += 32) {
    float xv = xr[c];
#pragma unroll
    for (int e = 0; e < NEXP; ++e) acc[e] += xv * wg_s[e * CDIM + c];
  }
#pragma unroll
  for (int e = 0; e < NEXP; ++e) {
    float v = acc[e];
    for (int off = 16; off > 0; off >>= 1) v += __shfl_down(v, off, 32);
    if (lane == 0) scores[(size_t)e * NTOK + n] = v;
  }
}

// ---------------------------------------------------------------------------
// 2) Per-expert top-1024 of 8192 (bitonic sort in LDS) + softmax over top vals
// ---------------------------------------------------------------------------
__global__ __launch_bounds__(1024) void topk_softmax_kernel(const float* __restrict__ scores,
                                                            int* __restrict__ token_idx,
                                                            float* __restrict__ token_w) {
  __shared__ float key[NTOK];   // 32 KB (negated scores -> ascending sort)
  __shared__ int   idx[NTOK];   // 32 KB
  const int e = blockIdx.x;
  const int tid = threadIdx.x;

  for (int i = tid; i < NTOK; i += 1024) {
    key[i] = -scores[(size_t)e * NTOK + i];
    idx[i] = i;
  }
  // bitonic sort (ascending on negated keys => descending on scores)
  for (int kk = 2; kk <= NTOK; kk <<= 1) {
    for (int jj = kk >> 1; jj > 0; jj >>= 1) {
      __syncthreads();
      for (int t = tid; t < NTOK / 2; t += 1024) {
        int i  = ((t / jj) * (jj << 1)) + (t % jj);
        int ix = i + jj;
        bool asc = ((i & kk) == 0);
        float ki = key[i], kj = key[ix];
        if ((ki > kj) == asc) {
          key[i] = kj; key[ix] = ki;
          int tmp = idx[i]; idx[i] = idx[ix]; idx[ix] = tmp;
        }
      }
    }
  }
  __syncthreads();

  // first CAP entries are the top-capacity scores
  float v   = -key[tid];
  int   tk  =  idx[tid];
  float mx  = -key[0];
  __syncthreads();
  float ev = __expf(v - mx);
  key[tid] = ev;
  __syncthreads();
  for (int s = 512; s > 0; s >>= 1) {
    if (tid < s) key[tid] += key[tid + s];
    __syncthreads();
  }
  float sum = key[0];
  token_w[e * CAP + tid]   = ev / sum;
  token_idx[e * CAP + tid] = tk;
}

// ---------------------------------------------------------------------------
// 3) Fused gather + SwiGLU up-projection:
//    h[e][m][n] = silu(xg.W1^T) * (xg.W3^T)     (bf16 out, f32 accumulate)
// ---------------------------------------------------------------------------
__global__ __launch_bounds__(256) void moe_swiglu_kernel(const float* __restrict__ x,
                                                         const float* __restrict__ W1,
                                                         const float* __restrict__ W3,
                                                         const int* __restrict__ token_idx,
                                                         unsigned short* __restrict__ hbuf) {
  __shared__ unsigned short As[BM * BK];
  __shared__ unsigned short B1s[BN * BK];
  __shared__ unsigned short B3s[BN * BK];
  __shared__ int tok_s[BM];

  const int e  = blockIdx.z;
  const int m0 = blockIdx.y * BM;
  const int n0 = blockIdx.x * BN;
  const int tid  = threadIdx.x;
  const int lane = tid & 31;
  const int wid  = tid >> 5;
  const int wm = wid & 3;   // 4 waves along M (32 rows each)
  const int wn = wid >> 2;  // 2 waves along N (64 cols each)
  const int g  = lane >> 4; // half-wave group
  const int nl = lane & 15;

  if (tid < BM) tok_s[tid] = token_idx[e * CAP + m0 + tid];

  const float* W1e = W1 + ((size_t)e * HDIM + n0) * CDIM;
  const float* W3e = W3 + ((size_t)e * HDIM + n0) * CDIM;

  v8f zero = {0.f, 0.f, 0.f, 0.f, 0.f, 0.f, 0.f, 0.f};
  v8f acc1[2][4], acc3[2][4];
#pragma unroll
  for (int i = 0; i < 2; ++i)
#pragma unroll
    for (int j = 0; j < 4; ++j) { acc1[i][j] = zero; acc3[i][j] = zero; }

  const int lr = tid & 7;   // 8 threads per 32-wide row (f32x4 each)
  const int r0 = tid >> 3;  // 32 rows per pass

  for (int k0 = 0; k0 < CDIM; k0 += BK) {
    __syncthreads();
#pragma unroll
    for (int p = 0; p < 4; ++p) {
      const int r = r0 + 32 * p;
      f32x4 a = *(const f32x4*)(x + (size_t)tok_s[r] * CDIM + k0 + lr * 4);
      u32x2 pa; pa.x = pk2(a.x, a.y); pa.y = pk2(a.z, a.w);
      *(u32x2*)&As[r * BK + lr * 4] = pa;
      f32x4 b1 = *(const f32x4*)(W1e + (size_t)r * CDIM + k0 + lr * 4);
      u32x2 p1; p1.x = pk2(b1.x, b1.y); p1.y = pk2(b1.z, b1.w);
      *(u32x2*)&B1s[r * BK + lr * 4] = p1;
      f32x4 b3 = *(const f32x4*)(W3e + (size_t)r * CDIM + k0 + lr * 4);
      u32x2 p3; p3.x = pk2(b3.x, b3.y); p3.y = pk2(b3.z, b3.w);
      *(u32x2*)&B3s[r * BK + lr * 4] = p3;
      if (k0 + BK < CDIM) {   // global_prefetch_b8 of next weight tiles
        __builtin_prefetch(W1e + (size_t)r * CDIM + k0 + BK + lr * 4, 0, 0);
        __builtin_prefetch(W3e + (size_t)r * CDIM + k0 + BK + lr * 4, 0, 0);
      }
    }
    __syncthreads();

    FragB af[2], b1f[4], b3f[4];
#pragma unroll
    for (int i = 0; i < 2; ++i) {
      const unsigned short* rp = &As[(32 * wm + 16 * i + nl) * BK];
      af[i].q[0] = *(const u32x4*)(rp + 16 * g);       // K = 16*g + 0..7
      af[i].q[1] = *(const u32x4*)(rp + 16 * g + 8);   // K = 16*g + 8..15
    }
#pragma unroll
    for (int j = 0; j < 4; ++j) {
      const unsigned short* rp1 = &B1s[(64 * wn + 16 * j + nl) * BK];
      b1f[j].q[0] = *(const u32x4*)(rp1 + 16 * g);     // K = 16*g + 0..7
      b1f[j].q[1] = *(const u32x4*)(rp1 + 16 * g + 8); // K = 16*g + 8..15
      const unsigned short* rp3 = &B3s[(64 * wn + 16 * j + nl) * BK];
      b3f[j].q[0] = *(const u32x4*)(rp3 + 16 * g);
      b3f[j].q[1] = *(const u32x4*)(rp3 + 16 * g + 8);
    }
#pragma unroll
    for (int i = 0; i < 2; ++i)
#pragma unroll
      for (int j = 0; j < 4; ++j) {
        acc1[i][j] = __builtin_amdgcn_wmma_f32_16x16x32_bf16(
            false, af[i].v, false, b1f[j].v, (short)0, acc1[i][j], false, false);
        acc3[i][j] = __builtin_amdgcn_wmma_f32_16x16x32_bf16(
            false, af[i].v, false, b3f[j].v, (short)0, acc3[i][j], false, false);
      }
  }

  // epilogue: silu(acc1) * acc3 -> bf16 h
#pragma unroll
  for (int i = 0; i < 2; ++i)
#pragma unroll
    for (int j = 0; j < 4; ++j)
#pragma unroll
      for (int r = 0; r < 8; ++r) {
        float a = acc1[i][j][r];
        float s = a / (1.f + __expf(-a));
        float hv = s * acc3[i][j][r];
        int m = m0 + 32 * wm + 16 * i + 8 * g + r;  // D: VGPR r -> M=r / M=r+8
        int n = n0 + 64 * wn + 16 * j + nl;         // N striped across lanes
        hbuf[((size_t)e * CAP + m) * HDIM + n] = f2bf(hv);
      }
}

// ---------------------------------------------------------------------------
// 4) Down projection + weighting + scatter-add:
//    out[tok[m]] += token_w[m] * (h[m] . W2^T)
//    A tile (already bf16) staged with global_load_async_to_lds when available.
// ---------------------------------------------------------------------------
__global__ __launch_bounds__(256) void moe_down_kernel(const unsigned short* __restrict__ hbuf,
                                                       const float* __restrict__ W2,
                                                       const int* __restrict__ token_idx,
                                                       const float* __restrict__ token_w,
                                                       float* __restrict__ out) {
  __shared__ unsigned short As[BM * BK];
  __shared__ unsigned short Bs[BN * BK];
  __shared__ int   tok_s[BM];
  __shared__ float tw_s[BM];

  const int e  = blockIdx.z;
  const int m0 = blockIdx.y * BM;
  const int n0 = blockIdx.x * BN;
  const int tid  = threadIdx.x;
  const int lane = tid & 31;
  const int wid  = tid >> 5;
  const int wm = wid & 3;
  const int wn = wid >> 2;
  const int g  = lane >> 4;
  const int nl = lane & 15;

  if (tid < BM) {
    tok_s[tid] = token_idx[e * CAP + m0 + tid];
    tw_s[tid]  = token_w[e * CAP + m0 + tid];
  }

  const unsigned short* Ae  = hbuf + ((size_t)e * CAP + m0) * HDIM;   // bf16 rows
  const float*          W2e = W2 + ((size_t)e * CDIM + n0) * HDIM;

  v8f zero = {0.f, 0.f, 0.f, 0.f, 0.f, 0.f, 0.f, 0.f};
  v8f acc[2][4];
#pragma unroll
  for (int i = 0; i < 2; ++i)
#pragma unroll
    for (int j = 0; j < 4; ++j) acc[i][j] = zero;

  const int la = tid & 3, ra = tid >> 2;  // A: 4 threads/row, 16B each
  const int lb = tid & 7, rb = tid >> 3;  // B: 8 threads/row, f32x4 each

  for (int k0 = 0; k0 < HDIM; k0 += BK) {
    __syncthreads();
#pragma unroll
    for (int p = 0; p < 2; ++p) {
      const int r = ra + 64 * p;
#if HAVE_ASYNC_LDS
      async_copy16(Ae + (size_t)r * HDIM + k0 + la * 8, &As[r * BK + la * 8]);
#else
      *(u32x4*)&As[r * BK + la * 8] =
          *(const u32x4*)(Ae + (size_t)r * HDIM + k0 + la * 8);
#endif
    }
#pragma unroll
    for (int p = 0; p < 4; ++p) {
      const int r = rb + 32 * p;
      f32x4 b = *(const f32x4*)(W2e + (size_t)r * HDIM + k0 + lb * 4);
      u32x2 pb; pb.x = pk2(b.x, b.y); pb.y = pk2(b.z, b.w);
      *(u32x2*)&Bs[r * BK + lb * 4] = pb;
      if (k0 + BK < HDIM)
        __builtin_prefetch(W2e + (size_t)r * HDIM + k0 + BK + lb * 4, 0, 0);
    }
#if HAVE_ASYNC_LDS
    wait_async0();   // own wave's async LDS writes done; barrier covers others
#endif
    __syncthreads();

    FragB af[2], bf[4];
#pragma unroll
    for (int i = 0; i < 2; ++i) {
      const unsigned short* rp = &As[(32 * wm + 16 * i + nl) * BK];
      af[i].q[0] = *(const u32x4*)(rp + 16 * g);
      af[i].q[1] = *(const u32x4*)(rp + 16 * g + 8);
    }
#pragma unroll
    for (int j = 0; j < 4; ++j) {
      const unsigned short* rp = &Bs[(64 * wn + 16 * j + nl) * BK];
      bf[j].q[0] = *(const u32x4*)(rp + 16 * g);
      bf[j].q[1] = *(const u32x4*)(rp + 16 * g + 8);
    }
#pragma unroll
    for (int i = 0; i < 2; ++i)
#pragma unroll
      for (int j = 0; j < 4; ++j)
        acc[i][j] = __builtin_amdgcn_wmma_f32_16x16x32_bf16(
            false, af[i].v, false, bf[j].v, (short)0, acc[i][j], false, false);
  }

  // epilogue: scale by token weight, scatter-add into out (tokens may repeat
  // across experts -> must use f32 atomic add)
#pragma unroll
  for (int i = 0; i < 2; ++i)
#pragma unroll
    for (int j = 0; j < 4; ++j)
#pragma unroll
      for (int r = 0; r < 8; ++r) {
        int ml = 32 * wm + 16 * i + 8 * g + r;
        int n  = n0 + 64 * wn + 16 * j + nl;
        float val = acc[i][j][r] * tw_s[ml];
        int token = tok_s[ml];
        __hip_atomic_fetch_add(&out[(size_t)token * CDIM + n], val,
                               __ATOMIC_RELAXED, __HIP_MEMORY_SCOPE_AGENT);
      }
}

// ---------------------------------------------------------------------------
extern "C" void kernel_launch(void* const* d_in, const int* in_sizes, int n_in,
                              void* d_out, int out_size, void* d_ws, size_t ws_size,
                              hipStream_t stream) {
  const float* x  = (const float*)d_in[0];  // [B,T,C]
  const float* Wg = (const float*)d_in[1];  // [E,C]
  const float* W1 = (const float*)d_in[2];  // [E,H,C]
  const float* W2 = (const float*)d_in[3];  // [E,C,H]
  const float* W3 = (const float*)d_in[4];  // [E,H,C]
  float* out = (float*)d_out;               // [B,T,C] f32

  // workspace layout (~34 MB)
  float* scores  = (float*)d_ws;                        // E*NTOK f32
  int*   tok_idx = (int*)(scores + NEXP * NTOK);        // E*CAP  i32
  float* tok_w   = (float*)(tok_idx + NEXP * CAP);      // E*CAP  f32
  unsigned short* hbuf = (unsigned short*)(tok_w + NEXP * CAP); // E*CAP*H bf16

  (void)hipMemsetAsync(d_out, 0, (size_t)out_size * sizeof(float), stream);

  gate_kernel<<<NTOK / 8, 256, 0, stream>>>(x, Wg, scores);
  topk_softmax_kernel<<<NEXP, 1024, 0, stream>>>(scores, tok_idx, tok_w);
  moe_swiglu_kernel<<<dim3(HDIM / BN, CAP / BM, NEXP), 256, 0, stream>>>(
      x, W1, W3, tok_idx, hbuf);
  moe_down_kernel<<<dim3(CDIM / BN, CAP / BM, NEXP), 256, 0, stream>>>(
      hbuf, W2, tok_idx, tok_w, out);
}
